// Model_Cond_BeT_26061861552752
// MI455X (gfx1250) — matrix operand order repaired and checked by reading, not compiled, not run on hardware
//
#include <hip/hip_runtime.h>
#include <stdint.h>

// Problem constants (match reference)
#define B_ROWS   32768
#define X_DIM    60
#define Y_DIM    10
#define N_K      64
#define HIDDEN   2048
#define OUT_DIM  704            // N_K + N_K*Y_DIM
#define IN_DIM   71             // Y_DIM + X_DIM + 1
#define K1PAD    96             // 71 padded to 3 * 32 for bf16 WMMA K-steps
#define MTILE    64             // rows per workgroup
#define HSTR     2056           // h row stride in bf16 elems (2048 + 8 pad)
#define NT2      (OUT_DIM/16)   // 44 N-tiles in stage 2

typedef __attribute__((ext_vector_type(16))) __bf16 v16bf;
typedef __attribute__((ext_vector_type(8)))  float  v8f;

union FragBF { uint4 q[2]; v16bf v; };
union AccF   { float f[8];  v8f   v; };

__global__ void k_zero_out(float* out) { out[0] = 0.0f; }

// W1 (71 x 2048, K-major)  ->  W1b (2048 x 96, N-major, K-contiguous, zero-padded)
__global__ void k_prep_w1(const float* __restrict__ W1, __bf16* __restrict__ W1b) {
  int total = HIDDEN * K1PAD;
  for (int idx = blockIdx.x * blockDim.x + threadIdx.x; idx < total; idx += gridDim.x * blockDim.x) {
    int n = idx / K1PAD, k = idx % K1PAD;
    float v = (k < IN_DIM) ? W1[k * HIDDEN + n] : 0.0f;
    W1b[idx] = (__bf16)v;                       // native v_cvt to bf16
  }
}

// W2 (2048 x 704) -> W2b (704 x 2048, N-major, K-contiguous)
__global__ void k_prep_w2(const float* __restrict__ W2, __bf16* __restrict__ W2b) {
  int total = OUT_DIM * HIDDEN;
  for (int idx = blockIdx.x * blockDim.x + threadIdx.x; idx < total; idx += gridDim.x * blockDim.x) {
    int n = idx / HIDDEN, k = idx % HIDDEN;
    W2b[idx] = (__bf16)W2[k * OUT_DIM + n];
  }
}

// Per-row nearest center + residual (independent of the MLP)
__global__ void k_labels(const float* __restrict__ yb, const float* __restrict__ centers,
                         int* __restrict__ labels, float* __restrict__ tres) {
  int i = blockIdx.x * blockDim.x + threadIdx.x;
  if (i >= B_ROWS) return;
  float y[Y_DIM];
  #pragma unroll
  for (int j = 0; j < Y_DIM; ++j) y[j] = yb[i * Y_DIM + j];
  int best = 0; float bestd = 3.4e38f;
  for (int c = 0; c < N_K; ++c) {
    float d = 0.0f;
    #pragma unroll
    for (int j = 0; j < Y_DIM; ++j) { float t = y[j] - centers[c * Y_DIM + j]; d += t * t; }
    if (d < bestd) { bestd = d; best = c; }
  }
  labels[i] = best;
  #pragma unroll
  for (int j = 0; j < Y_DIM; ++j) tres[i * Y_DIM + j] = y[j] - centers[best * Y_DIM + j];
}

// Fused: relu(x@W1+b1) @ W2 + b2  ->  CE + 100*MSE partial, atomic-reduced to out[0]
__global__ __launch_bounds__(256)
void k_fused(const float* __restrict__ x,
             const __bf16* __restrict__ W1b,
             const __bf16* __restrict__ W2b,
             const float* __restrict__ b1,
             const float* __restrict__ b2,
             const int* __restrict__ labels_g,
             const float* __restrict__ tres_g,
             float* __restrict__ out) {
  extern __shared__ char smem[];
  __bf16* hbuf = (__bf16*)smem;                                 // 64 x 2056 bf16 = 263168 B
  char* reg1 = smem + (size_t)MTILE * HSTR * 2;                 // 17408 B, shared A1/logits
  __bf16* A1    = (__bf16*)reg1;                                // stage1: 64 x 96 bf16
  float* logits = (float*)reg1;                                 // stage2: 64 x 68 f32
  float* stres = (float*)(reg1 + 17408);                        // 64 x 10 f32
  int*   slab  = (int*)((char*)stres + 2560);                   // 64 ints
  float* smse  = (float*)((char*)slab + 256);                   // 64 f32

  const int tid  = threadIdx.x;
  const int wave = tid >> 5;
  const int lane = tid & 31;
  const int l16  = lane & 15;
  const int half = lane >> 4;
  const int rowbase = blockIdx.x * MTILE;

  // --- Prologue: stage input tile (zero-padded [0|x|0]) + labels/residuals ---
  for (int idx = tid; idx < MTILE * K1PAD; idx += 256) {
    int m = idx / K1PAD, k = idx % K1PAD;
    float v = (k >= Y_DIM && k < Y_DIM + X_DIM) ? x[(rowbase + m) * X_DIM + (k - Y_DIM)] : 0.0f;
    A1[idx] = (__bf16)v;
  }
  for (int idx = tid; idx < MTILE; idx += 256) {
    slab[idx] = labels_g[rowbase + idx];
    smse[idx] = 0.0f;
  }
  for (int idx = tid; idx < MTILE * Y_DIM; idx += 256)
    stres[idx] = tres_g[rowbase * Y_DIM + idx];
  __syncthreads();

  // --- Stage 1: h(64 x 2048) = relu(A1 @ W1 + b1), bf16 into LDS ---
  {
    // Whole A tile register-resident: 4 mt x 3 ks fragments (96 VGPRs),
    // reused across all 16 N-tiles of this wave.
    FragBF afr[4][3];
    #pragma unroll
    for (int mt = 0; mt < 4; ++mt)
      #pragma unroll
      for (int ks = 0; ks < 3; ++ks) {
        int abase = (mt * 16 + l16) * K1PAD + ks * 32 + 8 * half;
        afr[mt][ks].q[0] = *(const uint4*)(A1 + abase);
        afr[mt][ks].q[1] = *(const uint4*)(A1 + abase + 16);
      }

    for (int ntl = 0; ntl < 16; ++ntl) {
      int nt   = wave * 16 + ntl;          // wave-uniform N tile (of 128)
      int ncol = nt * 16 + l16;
      const float bias = b1[ncol];
      AccF acc[4];
      #pragma unroll
      for (int mt = 0; mt < 4; ++mt)
        #pragma unroll
        for (int r = 0; r < 8; ++r) acc[mt].f[r] = 0.0f;

      #pragma unroll
      for (int ks = 0; ks < 3; ++ks) {
        FragBF b;
        int bbase = ncol * K1PAD + ks * 32 + 8 * half;
        b.q[0] = *(const uint4*)(W1b + bbase);
        b.q[1] = *(const uint4*)(W1b + bbase + 16);
        #pragma unroll
        for (int mt = 0; mt < 4; ++mt)      // 1 B load feeds 4 WMMAs
          acc[mt].v = __builtin_amdgcn_wmma_f32_16x16x32_bf16(false, afr[mt][ks].v, false, b.v,
                                                              (short)0, acc[mt].v, false, false);
      }
      #pragma unroll
      for (int mt = 0; mt < 4; ++mt)
        #pragma unroll
        for (int r = 0; r < 8; ++r) {
          int m = mt * 16 + r + 8 * half;
          float hv = fmaxf(acc[mt].f[r] + bias, 0.0f);
          hbuf[m * HSTR + ncol] = (__bf16)hv;   // single native cvt + b16 store
        }
    }
  }
  __syncthreads();

  // --- Stage 2: y = h @ W2 + b2; two N-tiles per wave so each A fragment
  //     (LDS) and its address math feed two WMMAs.  22 pairs over 8 waves. ---
  for (int p = wave; p < NT2 / 2; p += 8) {  // wave-uniform
    int ncol0 = (p * 2) * 16 + l16;          // 0..703
    int ncol1 = ncol0 + 16;
    const float bias0 = b2[ncol0];
    const float bias1 = b2[ncol1];
    AccF acc0[4], acc1[4];
    #pragma unroll
    for (int mt = 0; mt < 4; ++mt)
      #pragma unroll
      for (int r = 0; r < 8; ++r) { acc0[mt].f[r] = 0.0f; acc1[mt].f[r] = 0.0f; }

    #pragma unroll 2
    for (int ks = 0; ks < HIDDEN / 32; ++ks) {
      FragBF b0, b1f;
      int bb0 = ncol0 * HIDDEN + ks * 32 + 8 * half;
      int bb1 = ncol1 * HIDDEN + ks * 32 + 8 * half;
      b0.q[0]  = *(const uint4*)(W2b + bb0);   // L2-resident
      b0.q[1]  = *(const uint4*)(W2b + bb0 + 16);
      b1f.q[0] = *(const uint4*)(W2b + bb1);
      b1f.q[1] = *(const uint4*)(W2b + bb1 + 16);
      #pragma unroll
      for (int mt = 0; mt < 4; ++mt) {
        FragBF a;
        int abase = (mt * 16 + l16) * HSTR + ks * 32 + 8 * half;
        a.q[0] = *(const uint4*)(hbuf + abase);
        a.q[1] = *(const uint4*)(hbuf + abase + 16);
        acc0[mt].v = __builtin_amdgcn_wmma_f32_16x16x32_bf16(false, a.v, false, b0.v,
                                                             (short)0, acc0[mt].v, false, false);
        acc1[mt].v = __builtin_amdgcn_wmma_f32_16x16x32_bf16(false, a.v, false, b1f.v,
                                                             (short)0, acc1[mt].v, false, false);
      }
    }

    #pragma unroll
    for (int mt = 0; mt < 4; ++mt)
      #pragma unroll
      for (int r = 0; r < 8; ++r) {
        int m = mt * 16 + r + 8 * half;
        #pragma unroll
        for (int sel = 0; sel < 2; ++sel) {
          int   ncol = sel ? ncol1 : ncol0;
          float yv   = (sel ? acc1[mt].f[r] + bias1 : acc0[mt].f[r] + bias0);
          if (ncol < N_K) {
            logits[m * 68 + ncol] = yv;                     // stage-2 logit staging
          } else {
            int l = slab[m];
            int j = ncol - N_K - l * Y_DIM;                 // selected residual columns
            if (j >= 0 && j < Y_DIM) {
              float d = stres[m * Y_DIM + j] - yv;
              atomicAdd(&smse[m], d * d);                   // ds_add_f32
            }
          }
        }
      }
  }
  __syncthreads();

  // --- Epilogue: per-row softmax-CE + weighted MSE, one global atomic per row ---
  if (tid < MTILE) {
    int m = tid;
    float mx = -3.4e38f;
    for (int c = 0; c < N_K; ++c) mx = fmaxf(mx, logits[m * 68 + c]);
    float s = 0.0f;
    for (int c = 0; c < N_K; ++c) s += __expf(logits[m * 68 + c] - mx);
    int l = slab[m];
    float ce = -(logits[m * 68 + l] - mx - __logf(s));
    float contrib = ce * (1.0f / (float)B_ROWS)
                  + 100.0f * smse[m] * (1.0f / (float)(B_ROWS * Y_DIM));
    atomicAdd(out, contrib);                                // global_atomic_add_f32
  }
}

extern "C" void kernel_launch(void* const* d_in, const int* in_sizes, int n_in,
                              void* d_out, int out_size, void* d_ws, size_t ws_size,
                              hipStream_t stream) {
  (void)in_sizes; (void)n_in; (void)out_size; (void)ws_size;
  const float* x       = (const float*)d_in[0];
  const float* yb      = (const float*)d_in[1];
  const float* centers = (const float*)d_in[2];
  const float* W1      = (const float*)d_in[3];
  const float* b1      = (const float*)d_in[4];
  const float* W2      = (const float*)d_in[5];
  const float* b2      = (const float*)d_in[6];
  float* out = (float*)d_out;

  // Workspace layout (bf16 weights + labels + residuals), ~4.7 MB
  char* ws = (char*)d_ws;
  __bf16* W1b = (__bf16*)ws;                                        // 2048*96
  __bf16* W2b = (__bf16*)(ws + (size_t)HIDDEN * K1PAD * 2);         // 704*2048
  int*   labels = (int*)(ws + (size_t)HIDDEN * K1PAD * 2 + (size_t)OUT_DIM * HIDDEN * 2);
  float* tres   = (float*)((char*)labels + (size_t)B_ROWS * 4);

  k_zero_out<<<1, 1, 0, stream>>>(out);
  k_prep_w1<<<(HIDDEN * K1PAD + 255) / 256, 256, 0, stream>>>(W1, W1b);
  k_prep_w2<<<(OUT_DIM * HIDDEN + 255) / 256, 256, 0, stream>>>(W2, W2b);
  k_labels<<<(B_ROWS + 255) / 256, 256, 0, stream>>>(yb, centers, labels, tres);

  size_t smem = (size_t)MTILE * HSTR * 2   // h tile, bf16
              + 17408                      // A1 / logits (overlapped)
              + 2560 + 256 + 256;          // tres, labels, mse accumulators
  k_fused<<<B_ROWS / MTILE, 256, smem, stream>>>(x, W1b, W2b, b1, b2, labels, tres, out);
}
